// BitPackedLinear_77558519431325
// MI455X (gfx1250) — matrix operand order repaired and checked by reading, not compiled, not run on hardware
//
#include <hip/hip_runtime.h>
#include <stdint.h>

typedef __attribute__((ext_vector_type(16))) __bf16    v16bf;
typedef __attribute__((ext_vector_type(8)))  float     v8f;
typedef __attribute__((ext_vector_type(8)))  uint32_t  v8u;

#define M_TOK   4096
#define K_DIM   4096
#define N_DIM   4096
#define PBYTES  512          // packed bytes per output feature (4096/8)
#define BLK_M   128
#define BLK_N   256
#define KSTEP   64           // K per barrier (two 32-K WMMA sub-steps)
#define ROW_U32 36           // 144 B per LDS row: 64 bf16 data + pad (16B-aligned frags)

union Frag { v8u u; v16bf v; };

// {hi16(b), hi16(a)} in one v_perm_b32
__device__ __forceinline__ uint32_t pack_bf16_hi(float a, float b) {
  return __builtin_amdgcn_perm(__builtin_bit_cast(uint32_t, b),
                               __builtin_bit_cast(uint32_t, a), 0x07060302u);
}
__device__ __forceinline__ float bf16_trunc(float a) {
  return __builtin_bit_cast(float, __builtin_bit_cast(uint32_t, a) & 0xFFFF0000u);
}

// decode 16 packed weight bits -> 16 bf16 values in {-1,+1}
__device__ __forceinline__ v16bf decode_pm1(uint32_t bits) {
  Frag f;
  #pragma unroll
  for (int j = 0; j < 8; ++j)
    f.u[j] = 0xBF80BF80u ^ (((bits >> (2 * j)) & 1u) << 15)
                         ^ (((bits >> (2 * j + 1)) & 1u) << 31);
  return f.v;
}

__device__ __forceinline__ v16bf load_frag(const uint32_t* base) {
  uint4 p0 = *(const uint4*)(base);      // K = 8*half    + [0..7]
  uint4 p1 = *(const uint4*)(base + 8);  // K = 16+8*half + [0..7]
  Frag a;
  a.u[0] = p0.x; a.u[1] = p0.y; a.u[2] = p0.z; a.u[3] = p0.w;
  a.u[4] = p1.x; a.u[5] = p1.y; a.u[6] = p1.z; a.u[7] = p1.w;
  return a.v;
}

// stage 4 float4s (64 rows worth for this thread) as bf16 hi/lo planes
__device__ __forceinline__ void stage_half(const float4* v4, uint32_t* dstHi,
                                           uint32_t* dstLo, int srow, int scol,
                                           int rowbase) {
  #pragma unroll
  for (int i = 0; i < 4; ++i) {
    float4 v = v4[i];
    uint32_t h0 = pack_bf16_hi(v.x, v.y);
    uint32_t h1 = pack_bf16_hi(v.z, v.w);
    float lx = v.x - bf16_trunc(v.x);
    float ly = v.y - bf16_trunc(v.y);
    float lz = v.z - bf16_trunc(v.z);
    float lw = v.w - bf16_trunc(v.w);
    uint32_t l0 = pack_bf16_hi(lx, ly);
    uint32_t l1 = pack_bf16_hi(lz, lw);
    int off = (rowbase + srow + 16 * i) * ROW_U32 + scol * 2;
    uint2 th; th.x = h0; th.y = h1;
    uint2 tl; tl.x = l0; tl.y = l1;
    *(uint2*)(dstHi + off) = th;
    *(uint2*)(dstLo + off) = tl;
  }
}

#define WMMA_BF16(A, B, C) \
  __builtin_amdgcn_wmma_f32_16x16x32_bf16(false, (A), false, (B), (short)0, (C), false, false)

__global__ __launch_bounds__(256) void bitlinear_wmma(
    const float* __restrict__ x, const uint8_t* __restrict__ pw,
    const float* __restrict__ bias, float* __restrict__ out) {
  // double-buffered bf16 hi/lo planes of the 128x64 x-slab (72 KB total)
  __shared__ uint32_t ldsHi[2][BLK_M * ROW_U32];
  __shared__ uint32_t ldsLo[2][BLK_M * ROW_U32];

  const int tid   = threadIdx.x;
  const int lane  = tid & 31;
  const int wave  = tid >> 5;
  const int waveM = wave >> 2;      // 0..1 : 64 output rows each
  const int waveN = wave & 3;       // 0..3 : 64 output cols each
  const int lrow  = lane & 15;
  const int half  = lane >> 4;

  const int blockM = blockIdx.y * BLK_M;
  const int blockN = blockIdx.x * BLK_N;

  // staging: float4 id f = tid + i*256 ; row = f/16, col4 = f%16 (128x64 slab)
  const int srow = tid >> 4;        // 0..15 (+16 per i)
  const int scol = tid & 15;

  // single base pointer; per-i row deltas (16*K_DIM floats) fold into ioffsets
  const float* xrow = x + (size_t)(blockM + srow) * K_DIM + scol * 4;

  // B fragment (32x16, bf16): lane holds column N = lane%16, K = 8*(lane/16)+[0..7], 16+8*(lane/16)+[0..7]
  // one b64 load per N-tile per 64-K body; lane takes bytes half and 2+half of each word
  const uint8_t* pwb = pw + (size_t)(blockN + waveN * 64 + lrow) * PBYTES;

  v8f acc[4][4] = {};

  // prologue: load + stage slab 0
  {
    float4 c0[4], c1[4];
    #pragma unroll
    for (int i = 0; i < 4; ++i)
      c0[i] = *(const float4*)(xrow + (size_t)(16 * i) * K_DIM);
    #pragma unroll
    for (int i = 0; i < 4; ++i)
      c1[i] = *(const float4*)(xrow + (size_t)(16 * (i + 4)) * K_DIM);
    stage_half(c0, ldsHi[0], ldsLo[0], srow, scol, 0);
    stage_half(c1, ldsHi[0], ldsLo[0], srow, scol, 64);
  }
  __syncthreads();

  for (int k = 0; k < K_DIM; k += KSTEP) {
    const int  p    = (k >> 6) & 1;
    const bool more = (k + KSTEP) < K_DIM;
    const int  kn   = k + KSTEP;

    const uint32_t* hiB = ldsHi[p];
    const uint32_t* loB = ldsLo[p];
    uint32_t* dHi = ldsHi[p ^ 1];
    uint32_t* dLo = ldsLo[p ^ 1];

    // weight bits for both sub-steps: 4 x b64
    uint2 wb[4];
    #pragma unroll
    for (int nt = 0; nt < 4; ++nt)
      wb[nt] = *(const uint2*)(pwb + nt * 16 * PBYTES + (k >> 3));

    // prefetch rows 0..63 of next slab (32-WMMA latency shadow)
    float4 nxtA[4];
    if (more) {
      #pragma unroll
      for (int i = 0; i < 4; ++i)
        nxtA[i] = *(const float4*)(xrow + (size_t)(16 * i) * K_DIM + kn);
    }

    // ---- sub-step ks=0 : 32 WMMAs ----
    {
      v16bf b[4];
      #pragma unroll
      for (int nt = 0; nt < 4; ++nt)
        b[nt] = decode_pm1(((wb[nt].x >> (8 * half)) & 0xFFu) | (((wb[nt].x >> (16 + 8 * half)) & 0xFFu) << 8));
      #pragma unroll
      for (int mt = 0; mt < 4; ++mt) {
        const int rbase = (waveM * 64 + mt * 16 + lrow) * ROW_U32 + half * 4;
        v16bf ah = load_frag(hiB + rbase);
        v16bf al = load_frag(loB + rbase);
        #pragma unroll
        for (int nt = 0; nt < 4; ++nt)
          acc[mt][nt] = WMMA_BF16(ah, b[nt], acc[mt][nt]);
        #pragma unroll
        for (int nt = 0; nt < 4; ++nt)
          acc[mt][nt] = WMMA_BF16(al, b[nt], acc[mt][nt]);
      }
    }

    // stage first half; then prefetch rows 64..127 (ks=1 gives the shadow)
    float4 nxtB[4];
    if (more) {
      stage_half(nxtA, dHi, dLo, srow, scol, 0);
      #pragma unroll
      for (int i = 0; i < 4; ++i)
        nxtB[i] = *(const float4*)(xrow + (size_t)(16 * (i + 4)) * K_DIM + kn);
    }

    // ---- sub-step ks=1 : 32 WMMAs ----
    {
      v16bf b[4];
      #pragma unroll
      for (int nt = 0; nt < 4; ++nt)
        b[nt] = decode_pm1(((wb[nt].y >> (8 * half)) & 0xFFu) | (((wb[nt].y >> (16 + 8 * half)) & 0xFFu) << 8));
      #pragma unroll
      for (int mt = 0; mt < 4; ++mt) {
        const int rbase =
            (waveM * 64 + mt * 16 + lrow) * ROW_U32 + half * 4 + 16;
        v16bf ah = load_frag(hiB + rbase);
        v16bf al = load_frag(loB + rbase);
        #pragma unroll
        for (int nt = 0; nt < 4; ++nt)
          acc[mt][nt] = WMMA_BF16(ah, b[nt], acc[mt][nt]);
        #pragma unroll
        for (int nt = 0; nt < 4; ++nt)
          acc[mt][nt] = WMMA_BF16(al, b[nt], acc[mt][nt]);
      }
    }

    if (more) stage_half(nxtB, dHi, dLo, srow, scol, 64);
    __syncthreads();
  }

  // epilogue: C/D layout lane = (N=lane%16, M = 8*(lane/16)+vgpr)
  #pragma unroll
  for (int mt = 0; mt < 4; ++mt) {
    const int m0 = blockM + waveM * 64 + mt * 16 + half * 8;
    #pragma unroll
    for (int nt = 0; nt < 4; ++nt) {
      const int n = blockN + waveN * 64 + nt * 16 + lrow;
      const float bv = bias[n];
      #pragma unroll
      for (int v = 0; v < 8; ++v)
        out[(size_t)(m0 + v) * N_DIM + n] = acc[mt][nt][v] + bv;
    }
  }
}

extern "C" void kernel_launch(void* const* d_in, const int* in_sizes, int n_in,
                              void* d_out, int out_size, void* d_ws, size_t ws_size,
                              hipStream_t stream) {
  const float*   x    = (const float*)d_in[0];
  const uint8_t* pw   = (const uint8_t*)d_in[1];   // packed_weight, uint8 [4096,512]
  const float*   bias = (const float*)d_in[2];
  float*         out  = (float*)d_out;

  dim3 grid(N_DIM / BLK_N, M_TOK / BLK_M);  // (16, 32)
  bitlinear_wmma<<<grid, dim3(256), 0, stream>>>(x, pw, bias, out);
}
